// FeatureFuser_44727789421256
// MI455X (gfx1250) — matrix operand order repaired and checked by reading, not compiled, not run on hardware
//
#include <hip/hip_runtime.h>
#include <stdint.h>

// ---------------- model constants ----------------
#define E_DIM 4096
#define KVH_HEADS 8
#define HD_DIM 128
#define G_GRP 4
#define B_SZ 32

// ---------------- gemm tiling ----------------
#define BM 128
#define BN 128
#define BK 32
#define LDS_STRIDE 40   // BK + 8 bf16 pad -> 80B rows, keeps 16B alignment

typedef __attribute__((ext_vector_type(16))) __bf16 v16bf;
typedef __attribute__((ext_vector_type(8)))  float  v8f;

// native f32 -> bf16 (RNE) conversion; lowers to v_cvt_*bf16* on gfx1250
__device__ __forceinline__ unsigned short f2bf(float f) {
  __bf16 h = (__bf16)f;
  return __builtin_bit_cast(unsigned short, h);
}
__device__ __forceinline__ float bf2f(unsigned short h) {
  unsigned int u = ((unsigned int)h) << 16;
  return __builtin_bit_cast(float, u);
}

struct GemmArgs {
  const unsigned short* A;   // bf16 activations, row-major, k-contiguous
  const float*          B;   // f32 "weight" operand: element [n,k] at n*ldbN + k*ldbK
  const float*          bias;
  float*                C;
  unsigned short*       Cbf;
  int M, N, K;
  int lda, ldbN, ldbK, ldc;
  long long sA1, sA0, sB1, sB0, sC1, sC0;   // batch strides, z = z1*z0count + z0
  int z0count;
  float alpha;
};

// C = alpha * (A @ B^T) + bias ; A bf16, B f32 converted to bf16 in LDS.
// A tiles are staged with GLOBAL_LOAD_ASYNC_TO_LDS_B128 (ASYNCcnt) when fully
// in-bounds; B tiles go through VGPRs for the f32->bf16 conversion.
template <bool BTRANS>
__global__ __launch_bounds__(256)
void gemm_bf16_wmma(GemmArgs g) {
  __shared__ unsigned short As[2][BM][LDS_STRIDE];
  __shared__ unsigned short Bs[2][BN][LDS_STRIDE];

  const int tid  = threadIdx.x;
  const int lane = tid & 31;
  const int wave = tid >> 5;
  const int wm   = wave >> 2;   // 0..1 : 64-row half
  const int wn   = wave & 3;    // 0..3 : 32-col quarter

  const int z  = blockIdx.z;
  const int z1 = z / g.z0count;
  const int z0 = z - z1 * g.z0count;
  const unsigned short* Ab = g.A + (long long)z1 * g.sA1 + (long long)z0 * g.sA0;
  const float*          Bb = g.B + (long long)z1 * g.sB1 + (long long)z0 * g.sB0;
  const long long cOff = (long long)z1 * g.sC1 + (long long)z0 * g.sC0;

  const int m0 = blockIdx.y * BM;
  const int n0 = blockIdx.x * BN;

  v8f acc[4][2] = {};
  uint4 aReg[2];
  uint4 bReg[2][2];

  const int numK = (g.K + BK - 1) / BK;

  // ---- A staging: async copy straight into LDS for full tiles ----
  auto issueA = [&](int k0, int bufSel) -> bool {
    if (m0 + BM <= g.M && k0 + BK <= g.K) {
#pragma unroll
      for (int i = 0; i < 2; ++i) {
        const int c  = tid + i * 256;
        const int r  = c >> 2;
        const int kc = (c & 3) << 3;
        const unsigned short* ga = Ab + (long long)(m0 + r) * g.lda + (k0 + kc);
        const unsigned ldsa = (unsigned)(uintptr_t)&As[bufSel][r][kc];
        asm volatile("global_load_async_to_lds_b128 %0, %1, off"
                     :: "v"(ldsa), "v"(ga) : "memory");
      }
      return true;
    }
    // edge tile: register path with zero-padding
#pragma unroll
    for (int i = 0; i < 2; ++i) {
      const int c  = tid + i * 256;
      const int r  = c >> 2;
      const int kc = (c & 3) << 3;
      const int gm = m0 + r;
      const int gk = k0 + kc;
      if (gm < g.M && gk + 8 <= g.K) {
        aReg[i] = *(const uint4*)(Ab + (long long)gm * g.lda + gk);
      } else {
        union { unsigned short h[8]; uint4 q; } t;
#pragma unroll
        for (int e = 0; e < 8; ++e)
          t.h[e] = (gm < g.M && gk + e < g.K)
                     ? Ab[(long long)gm * g.lda + gk + e] : (unsigned short)0;
        aReg[i] = t.q;
      }
    }
    return false;
  };

  auto lstoreA = [&](int bufSel) {
#pragma unroll
    for (int i = 0; i < 2; ++i) {
      const int c  = tid + i * 256;
      const int r  = c >> 2;
      const int kc = (c & 3) << 3;
      *(uint4*)&As[bufSel][r][kc] = aReg[i];
    }
  };

  // ---- B staging: registers (needs f32->bf16 convert) ----
  auto gloadB = [&](int k0) {
#pragma unroll
    for (int i = 0; i < 2; ++i) {
      const int c  = tid + i * 256;
      const int r  = c >> 2;
      const int kc = (c & 3) << 3;
      const int gn = n0 + r;
      const int gk = k0 + kc;
      if (!BTRANS) {
        if (gn < g.N && gk + 8 <= g.K) {
          const uint4* pb = (const uint4*)(Bb + (long long)gn * g.ldbN + gk);
          bReg[i][0] = pb[0];
          bReg[i][1] = pb[1];
          __builtin_prefetch(Bb + (long long)gn * g.ldbN + gk + 2 * BK, 0, 1);
        } else {
          union { float f[8]; uint4 q[2]; } t;
#pragma unroll
          for (int e = 0; e < 8; ++e)
            t.f[e] = (gn < g.N && gk + e < g.K)
                       ? Bb[(long long)gn * g.ldbN + gk + e] : 0.0f;
          bReg[i][0] = t.q[0]; bReg[i][1] = t.q[1];
        }
      } else {
        union { float f[8]; uint4 q[2]; } t;
#pragma unroll
        for (int e = 0; e < 8; ++e)
          t.f[e] = (gn < g.N && gk + e < g.K)
                     ? Bb[(long long)gn * g.ldbN + (long long)(gk + e) * g.ldbK] : 0.0f;
        bReg[i][0] = t.q[0]; bReg[i][1] = t.q[1];
      }
    }
  };

  auto lstoreB = [&](int bufSel) {
#pragma unroll
    for (int i = 0; i < 2; ++i) {
      const int c  = tid + i * 256;
      const int r  = c >> 2;
      const int kc = (c & 3) << 3;
      union { uint4 q[2]; float f[8]; } t;
      t.q[0] = bReg[i][0]; t.q[1] = bReg[i][1];
      union { unsigned short h[8]; uint4 q; } o;
#pragma unroll
      for (int e = 0; e < 8; ++e) o.h[e] = f2bf(t.f[e]);
      *(uint4*)&Bs[bufSel][r][kc] = o.q;
    }
  };

  bool aAsync = issueA(0, 0);
  gloadB(0);
  int buf = 0;
  for (int kt = 0; kt < numK; ++kt) {
    if (!aAsync) lstoreA(buf);
    lstoreB(buf);
    asm volatile("s_wait_asynccnt 0x0" ::: "memory");
    __syncthreads();
    bool nextAsync = false;
    if (kt + 1 < numK) {
      nextAsync = issueA((kt + 1) * BK, buf ^ 1);
      gloadB((kt + 1) * BK);
    }

    // fragment gather per ISA 16-bit A(16x32) / B(32x16) wave32 layouts
    v16bf af[4], bf[2];
    {
      const int ar  = lane & 15;
      const int akb = (lane < 16) ? 0 : 8;
#pragma unroll
      for (int mi = 0; mi < 4; ++mi) {
        const int row = wm * 64 + mi * 16 + ar;
        union { uint4 q[2]; v16bf v; } t;
        t.q[0] = *(const uint4*)&As[buf][row][akb];
        t.q[1] = *(const uint4*)&As[buf][row][akb + 16];
        af[mi] = t.v;
      }
      const int bn  = lane & 15;
      const int bkb = (lane < 16) ? 0 : 8;
#pragma unroll
      for (int ni = 0; ni < 2; ++ni) {
        const int col = wn * 32 + ni * 16 + bn;
        union { uint4 q[2]; v16bf v; } t;
        t.q[0] = *(const uint4*)&Bs[buf][col][bkb];
        t.q[1] = *(const uint4*)&Bs[buf][col][bkb + 16];
        bf[ni] = t.v;
      }
    }
#pragma unroll
    for (int mi = 0; mi < 4; ++mi)
#pragma unroll
      for (int ni = 0; ni < 2; ++ni)
        acc[mi][ni] = __builtin_amdgcn_wmma_f32_16x16x32_bf16(
            false, af[mi], false, bf[ni], (short)0, acc[mi][ni], false, false);
    aAsync = nextAsync;
    buf ^= 1;
  }

  // epilogue: C layout lanes 0-15 -> rows 0..7, lanes 16-31 -> rows 8..15
  const int cn   = lane & 15;
  const int rsel = (lane < 16) ? 0 : 8;
#pragma unroll
  for (int mi = 0; mi < 4; ++mi) {
#pragma unroll
    for (int ni = 0; ni < 2; ++ni) {
      const int gn = n0 + wn * 32 + ni * 16 + cn;
      if (gn >= g.N) continue;
      const float bv = g.bias ? g.bias[gn] : 0.0f;
#pragma unroll
      for (int v = 0; v < 8; ++v) {
        const int gm = m0 + wm * 64 + mi * 16 + rsel + v;
        if (gm >= g.M) continue;
        const float val = acc[mi][ni][v] * g.alpha + bv;
        const long long off = cOff + (long long)gm * g.ldc + gn;
        if (g.C)   g.C[off]   = val;
        if (g.Cbf) g.Cbf[off] = f2bf(val);
      }
    }
  }
}

// ---------------- layernorm (cols == 4096) ----------------
__global__ __launch_bounds__(256)
void ln_kernel(const float* xf, const unsigned short* xh, const float* resid,
               const float* gamma, const float* beta,
               float* outf, unsigned short* outh,
               int cols, int rows_per_batch, long long out_batch_stride, int out_ld)
{
  __shared__ float red[256];
  const int row = blockIdx.x;
  const int tid = threadIdx.x;
  const long long base = (long long)row * cols;
  float vals[16];
  int cnt = 0;
  float s = 0.f;
  for (int i = tid; i < cols; i += 256) {
    float x = xf ? xf[base + i] : bf2f(xh[base + i]);
    if (resid) x += resid[base + i];
    vals[cnt++] = x;
    s += x;
  }
  red[tid] = s; __syncthreads();
  for (int o = 128; o > 0; o >>= 1) { if (tid < o) red[tid] += red[tid + o]; __syncthreads(); }
  const float mean = red[0] / cols;
  __syncthreads();
  float vs = 0.f;
  for (int j = 0; j < cnt; ++j) { const float d = vals[j] - mean; vs += d * d; }
  red[tid] = vs; __syncthreads();
  for (int o = 128; o > 0; o >>= 1) { if (tid < o) red[tid] += red[tid + o]; __syncthreads(); }
  const float inv = rsqrtf(red[0] / cols + 1e-5f);
  const long long ob = (long long)(row / rows_per_batch) * out_batch_stride
                     + (long long)(row % rows_per_batch) * out_ld;
  cnt = 0;
  for (int i = tid; i < cols; i += 256) {
    const float y = (vals[cnt++] - mean) * inv * gamma[i] + beta[i];
    if (outf) outf[ob + i] = y;
    if (outh) outh[ob + i] = f2bf(y);
  }
}

// ---------------- key-padding mask ----------------
__global__ __launch_bounds__(256)
void mask_kernel(const float* feat, int D, unsigned char* mask)
{
  __shared__ int any;
  const int row = blockIdx.x;
  if (threadIdx.x == 0) any = 0;
  __syncthreads();
  int local = 0;
  const long long base = (long long)row * D;
  for (int i = threadIdx.x; i < D; i += 256)
    if (feat[base + i] != 0.0f) { local = 1; break; }
  if (local) atomicOr(&any, 1);
  __syncthreads();
  if (threadIdx.x == 0) mask[row] = (unsigned char)(any ? 0 : 1);
}

// ---------------- masked softmax over rows (len <= 1024), writes bf16 P ----
__global__ __launch_bounds__(256)
void softmax_kernel(const float* S, unsigned short* P, int len,
                    const unsigned char* mask, int rows_per_b, int mask_stride)
{
  __shared__ float red[256];
  const int row = blockIdx.x;
  const int tid = threadIdx.x;
  const long long base = (long long)row * len;
  const unsigned char* mrow = mask
      ? (mask + (long long)(row / rows_per_b) * mask_stride) : nullptr;
  float v[4];
  int cnt = 0;
  float mx = -3.4e38f;
  for (int i = tid; i < len; i += 256) {
    float x = S[base + i];
    if (mrow && mrow[i]) x = -3.4e38f;
    v[cnt++] = x;
    mx = fmaxf(mx, x);
  }
  red[tid] = mx; __syncthreads();
  for (int o = 128; o > 0; o >>= 1) { if (tid < o) red[tid] = fmaxf(red[tid], red[tid + o]); __syncthreads(); }
  mx = red[0]; __syncthreads();
  float s = 0.f;
  for (int j = 0; j < cnt; ++j) { v[j] = __expf(v[j] - mx); s += v[j]; }
  red[tid] = s; __syncthreads();
  for (int o = 128; o > 0; o >>= 1) { if (tid < o) red[tid] += red[tid + o]; __syncthreads(); }
  const float inv = (red[0] > 0.f) ? 1.0f / red[0] : 0.f;
  cnt = 0;
  for (int i = tid; i < len; i += 256)
    P[base + i] = f2bf(v[cnt++] * inv);
}

// qproj (BQ,nq,4096) f32 -> (BQ,KVH,R=G*nq,HD) bf16
__global__ void rearrange_q_kernel(const float* qp, unsigned short* qr,
                                   int nq, long long total)
{
  const long long idx = (long long)blockIdx.x * 256 + threadIdx.x;
  if (idx >= total) return;
  const int col = (int)(idx & 4095);
  const long long t = idx >> 12;
  const int qi = (int)(t % nq);
  const long long bq = t / nq;
  const int kvh = col >> 9;
  const int gg  = (col >> 7) & 3;
  const int d   = col & 127;
  const int R   = G_GRP * nq;
  const long long o = (((bq * KVH_HEADS + kvh) * R + (long long)gg * nq + qi)) * HD_DIM + d;
  qr[o] = f2bf(qp[idx]);
}

// O (B,KVH,R,HD) f32 -> (B,nq,E) bf16
__global__ void rearrange_o_kernel(const float* O, unsigned short* orh,
                                   int nq, long long total)
{
  const long long idx = (long long)blockIdx.x * 256 + threadIdx.x;
  if (idx >= total) return;
  const int d = (int)(idx & 127);
  long long t = idx >> 7;
  const int R = G_GRP * nq;
  const int r = (int)(t % R);
  t /= R;
  const int kvh = (int)(t % KVH_HEADS);
  const long long b = t / KVH_HEADS;
  const int gg = r / nq, qi = r % nq;
  const long long o = (b * nq + qi) * E_DIM + (long long)kvh * (G_GRP * HD_DIM)
                      + (long long)gg * HD_DIM + d;
  orh[o] = f2bf(O[idx]);
}

// f32 rows (rows,D) -> bf16 rows (rows,Dpad), pad zero-filled
__global__ void f32_to_bf16_rows_kernel(const float* x, unsigned short* y,
                                        int D, int Dpad, long long total)
{
  const long long i = (long long)blockIdx.x * 256 + threadIdx.x;
  if (i >= total) return;
  const long long r = i / Dpad;
  const int c = (int)(i - r * Dpad);
  y[i] = (c < D) ? f2bf(x[r * D + c]) : (unsigned short)0;
}

// copy (B,8,4096) bf16 into tokens (B,48,4096) bf16 at row_off
__global__ void copy_tokens_kernel(const unsigned short* src, unsigned short* dst,
                                   int rows_src_per_b, int row_off, long long total)
{
  const long long i = (long long)blockIdx.x * 256 + threadIdx.x;
  if (i >= total) return;
  const int col = (int)(i & 4095);
  const long long t = i >> 12;
  const int q = (int)(t % rows_src_per_b);
  const long long b = t / rows_src_per_b;
  dst[((b * 48 + row_off + q) * (long long)E_DIM) + col] = src[i];
}

// ================= host side =================
struct GqaW { const float *bo, *wk, *wo, *wq, *wv; };
struct Scr  { unsigned short* QR; float* QP; float* S; unsigned short* P;
              float* O; unsigned short* OR_; float* K; float* V; };

static void launch_gemm(hipStream_t st, bool btrans,
  const unsigned short* A, const float* Bp, const float* bias,
  float* C, unsigned short* Cbf,
  int M, int N, int K, int lda, int ldbN, int ldbK, int ldc,
  long long sA1, long long sA0, long long sB1, long long sB0,
  long long sC1, long long sC0, int Z, int z0count, float alpha)
{
  GemmArgs a;
  a.A = A; a.B = Bp; a.bias = bias; a.C = C; a.Cbf = Cbf;
  a.M = M; a.N = N; a.K = K;
  a.lda = lda; a.ldbN = ldbN; a.ldbK = ldbK; a.ldc = ldc;
  a.sA1 = sA1; a.sA0 = sA0; a.sB1 = sB1; a.sB0 = sB0; a.sC1 = sC1; a.sC0 = sC0;
  a.z0count = z0count; a.alpha = alpha;
  dim3 grid((N + BN - 1) / BN, (M + BM - 1) / BM, Z);
  if (btrans) gemm_bf16_wmma<true ><<<grid, 256, 0, st>>>(a);
  else        gemm_bf16_wmma<false><<<grid, 256, 0, st>>>(a);
}

static void run_gqa(hipStream_t st, int B, int BQ, int Lq, int Lk,
                    const unsigned short* q_in, const unsigned short* kv_in,
                    const GqaW& w, const unsigned char* mask,
                    float* outC, unsigned short* outCbf, long long sC0out,
                    const Scr& s)
{
  const int R = G_GRP * Lq;
  // q / k / v projections
  launch_gemm(st, false, q_in, w.wq, nullptr, s.QP, nullptr,
              BQ * Lq, E_DIM, E_DIM, E_DIM, E_DIM, 1, E_DIM,
              0, 0, 0, 0, 0, 0, 1, 1, 1.0f);
  launch_gemm(st, false, kv_in, w.wk, nullptr, s.K, nullptr,
              B * Lk, KVH_HEADS * HD_DIM, E_DIM, E_DIM, E_DIM, 1, KVH_HEADS * HD_DIM,
              0, 0, 0, 0, 0, 0, 1, 1, 1.0f);
  launch_gemm(st, false, kv_in, w.wv, nullptr, s.V, nullptr,
              B * Lk, KVH_HEADS * HD_DIM, E_DIM, E_DIM, E_DIM, 1, KVH_HEADS * HD_DIM,
              0, 0, 0, 0, 0, 0, 1, 1, 1.0f);
  // rearrange q into per-kv-head tiles
  { const long long tot = (long long)BQ * Lq * E_DIM;
    rearrange_q_kernel<<<(unsigned)((tot + 255) / 256), 256, 0, st>>>(s.QP, s.QR, Lq, tot); }
  // scores: per z=(b,kvh): S(R,Lk) = (1/sqrt(HD)) * Qr @ K^T
  const long long sA1 = (BQ == 1) ? 0 : (long long)KVH_HEADS * R * HD_DIM;
  launch_gemm(st, false, s.QR, s.K, nullptr, s.S, nullptr,
              R, Lk, HD_DIM,
              HD_DIM, KVH_HEADS * HD_DIM, 1, Lk,
              sA1, (long long)R * HD_DIM,
              (long long)Lk * KVH_HEADS * HD_DIM, HD_DIM,
              (long long)KVH_HEADS * R * Lk, (long long)R * Lk,
              B * KVH_HEADS, KVH_HEADS, 0.08838834764831845f);
  // softmax (+ key padding mask) -> bf16 P
  softmax_kernel<<<B * KVH_HEADS * R, 256, 0, st>>>(s.S, s.P, Lk, mask,
                                                    KVH_HEADS * R, Lk);
  // O = P @ V  (B operand is V^T: ldbN=1, ldbK=1024)
  launch_gemm(st, true, s.P, s.V, nullptr, s.O, nullptr,
              R, HD_DIM, Lk,
              Lk, 1, KVH_HEADS * HD_DIM, HD_DIM,
              (long long)KVH_HEADS * R * Lk, (long long)R * Lk,
              (long long)Lk * KVH_HEADS * HD_DIM, HD_DIM,
              (long long)KVH_HEADS * R * HD_DIM, (long long)R * HD_DIM,
              B * KVH_HEADS, KVH_HEADS, 1.0f);
  // heads -> (B, Lq, E) bf16
  { const long long tot = (long long)B * KVH_HEADS * R * HD_DIM;
    rearrange_o_kernel<<<(unsigned)((tot + 255) / 256), 256, 0, st>>>(s.O, s.OR_, Lq, tot); }
  // out = O @ wo^T + bo, batched over b for flexible C placement
  launch_gemm(st, false, s.OR_, w.wo, w.bo, outC, outCbf,
              Lq, E_DIM, E_DIM, E_DIM, E_DIM, 1, E_DIM,
              0, (long long)Lq * E_DIM, 0, 0, 0, sC0out,
              B, B, 1.0f);
}

extern "C" void kernel_launch(void* const* d_in, const int* in_sizes, int n_in,
                              void* d_out, int out_size, void* d_ws, size_t ws_size,
                              hipStream_t stream)
{
  (void)in_sizes; (void)n_in; (void)out_size; (void)ws_size;
  auto F = [&](int i) { return (const float*)d_in[i]; };
  // flatten order (alphabetical dict keys, jax pytree):
  // 0 feat_2d, 1 feat_3d, 2 feat_protein, 3 final_b, 4 final_w,
  // fuser[L] at 5+L*14: a23{bo,wk,wo,wq,wv}, a32{bo,wk,wo,wq,wv}, n2b,n2g,n3b,n3g
  // ql2d@33, ql3d@45, qlp@57: attn{bo,wk,wo,wq,wv}, bin, lnkv_b, lnkv_g,
  //                           lnq_b, lnq_g, query, win
  const float* feat2d  = F(0);
  const float* feat3d  = F(1);
  const float* featp   = F(2);
  const float* final_b = F(3);
  const float* final_w = F(4);

  const int B = B_SZ;
  char* ws = (char*)d_ws;
  size_t off = 0;
  auto alloc = [&](size_t bytes) -> void* {
    void* p = ws + off; off = (off + bytes + 255) & ~(size_t)255; return p;
  };

  unsigned short* FB   = (unsigned short*)alloc((size_t)B * 1024 * 1280 * 2);
  unsigned short* XB   = (unsigned short*)alloc((size_t)B * 1024 * 4096 * 2);
  unsigned short* KVB  = (unsigned short*)alloc((size_t)B * 1024 * 4096 * 2);
  float*          KBUF = (float*)alloc((size_t)B * 1024 * 1024 * 4);
  float*          VBUF = (float*)alloc((size_t)B * 1024 * 1024 * 4);
  float*          QPROJ= (float*)alloc((size_t)B * 8 * 4096 * 4);
  unsigned short* QR   = (unsigned short*)alloc((size_t)B * KVH_HEADS * 32 * 128 * 2);
  float*          SBUF = (float*)XB;  // reuse: XB dead once KVB is produced
  unsigned short* PBUF = (unsigned short*)alloc((size_t)B * KVH_HEADS * 128 * 1024 * 2);
  float*          OBUF = (float*)alloc((size_t)B * KVH_HEADS * 128 * 128 * 4);
  unsigned short* ORB  = (unsigned short*)alloc((size_t)B * 32 * 4096 * 2);
  unsigned char*  MASKB= (unsigned char*)alloc((size_t)B * 1024);
  unsigned short* QLN  = (unsigned short*)alloc((size_t)32 * 4096 * 2);
  float*          T2F  = (float*)alloc((size_t)B * 8 * 4096 * 4);
  unsigned short* T2H  = (unsigned short*)alloc((size_t)B * 8 * 4096 * 2);
  float*          T3F  = (float*)alloc((size_t)B * 8 * 4096 * 4);
  unsigned short* T3H  = (unsigned short*)alloc((size_t)B * 8 * 4096 * 2);
  float*          TMPA = (float*)alloc((size_t)B * 8 * 4096 * 4);
  unsigned short* TOK  = (unsigned short*)alloc((size_t)B * 48 * 4096 * 2);

  const Scr scr{QR, QPROJ, SBUF, PBUF, OBUF, ORB, KBUF, VBUF};

  auto run_ql = [&](const float* feat, int Dk, int L, int nq, int base,
                    float* outC, unsigned short* outCbf, long long sC0out) {
    const GqaW w = { F(base + 0), F(base + 1), F(base + 2), F(base + 3), F(base + 4) };
    const float* bin   = F(base + 5);
    const float* lnkvb = F(base + 6);
    const float* lnkvg = F(base + 7);
    const float* lnqb  = F(base + 8);
    const float* lnqg  = F(base + 9);
    const float* query = F(base + 10);
    const float* win   = F(base + 11);
    const long long rowsKV = (long long)B * L;
    const int Dpad = (Dk + 7) & ~7;   // keep bf16 rows 16B aligned

    mask_kernel<<<(unsigned)rowsKV, 256, 0, stream>>>(feat, Dk, MASKB);
    { const long long tot = rowsKV * Dpad;
      f32_to_bf16_rows_kernel<<<(unsigned)((tot + 255) / 256), 256, 0, stream>>>(
          feat, FB, Dk, Dpad, tot); }
    // x = feat @ win^T + bin  (bf16 only)
    launch_gemm(stream, false, FB, win, bin, nullptr, XB,
                (int)rowsKV, E_DIM, Dk, Dpad, Dk, 1, E_DIM,
                0, 0, 0, 0, 0, 0, 1, 1, 1.0f);
    // kv = LN(x)
    ln_kernel<<<(unsigned)rowsKV, 256, 0, stream>>>(nullptr, XB, nullptr,
        lnkvg, lnkvb, nullptr, KVB, E_DIM, (int)rowsKV, 0, E_DIM);
    // q = LN(query)
    ln_kernel<<<nq, 256, 0, stream>>>(query, nullptr, nullptr,
        lnqg, lnqb, nullptr, QLN, E_DIM, nq, 0, E_DIM);
    run_gqa(stream, B, /*BQ=*/1, nq, L, QLN, KVB, w, MASKB,
            outC, outCbf, sC0out, scr);
  };

  run_ql(feat2d, 300, 256, 8, 33, T2F, T2H, (long long)8 * E_DIM);
  run_ql(feat3d, 512, 256, 8, 45, T3F, T3H, (long long)8 * E_DIM);
  run_ql(featp, 1280, 1024, 32, 57, nullptr, TOK + (long long)16 * E_DIM,
         (long long)48 * E_DIM);

  // fuser: f2 = LN(f2 + gqa(f2,f3,f3)); f3 = LN(f3 + gqa(f3,f2,f2))
  for (int L = 0; L < 2; ++L) {
    const int bb = 5 + L * 14;
    const GqaW w23 = { F(bb + 0), F(bb + 1), F(bb + 2), F(bb + 3), F(bb + 4) };
    const GqaW w32 = { F(bb + 5), F(bb + 6), F(bb + 7), F(bb + 8), F(bb + 9) };
    const float* n2b = F(bb + 10); const float* n2g = F(bb + 11);
    const float* n3b = F(bb + 12); const float* n3g = F(bb + 13);

    run_gqa(stream, B, /*BQ=*/B, 8, 8, T2H, T3H, w23, nullptr,
            TMPA, nullptr, (long long)8 * E_DIM, scr);
    ln_kernel<<<B * 8, 256, 0, stream>>>(T2F, nullptr, TMPA, n2g, n2b,
        T2F, T2H, E_DIM, B * 8, 0, E_DIM);

    run_gqa(stream, B, /*BQ=*/B, 8, 8, T3H, T2H, w32, nullptr,
            TMPA, nullptr, (long long)8 * E_DIM, scr);
    ln_kernel<<<B * 8, 256, 0, stream>>>(T3F, nullptr, TMPA, n3g, n3b,
        T3F, T3H, E_DIM, B * 8, 0, E_DIM);
  }

  // assemble tokens (B,48,4096) bf16
  { const long long tot = (long long)B * 8 * E_DIM;
    copy_tokens_kernel<<<(unsigned)((tot + 255) / 256), 256, 0, stream>>>(T2H, TOK, 8, 0, tot);
    copy_tokens_kernel<<<(unsigned)((tot + 255) / 256), 256, 0, stream>>>(T3H, TOK, 8, 8, tot);
  }

  // final head: out = tokens @ final_w^T + final_b (f32 output)
  launch_gemm(stream, false, TOK, final_w, final_b, (float*)d_out, nullptr,
              B * 48, E_DIM, E_DIM, E_DIM, E_DIM, 1, E_DIM,
              0, 0, 0, 0, 0, 0, 1, 1, 1.0f);
}
